// HamiltonianFlow_50319836840539
// MI455X (gfx1250) — matrix operand order repaired and checked by reading, not compiled, not run on hardware
//
#include <hip/hip_runtime.h>
#include <math.h>

typedef __attribute__((ext_vector_type(16))) _Float16 v16h;
typedef __attribute__((ext_vector_type(8)))  float    v8f;

#define NBATCH 256
#define FEAT   256
#define HID    256
#define TILE_M 16
#define NWAVE  16
#define BLOCK  (NWAVE * 32)
#define STEPS  100
#define DT     0.01f
#define PITCH  (FEAT + 8)   // padded LDS row pitch (halfs); 528B keeps b128 alignment

#if __has_builtin(__builtin_amdgcn_tanhf)
#define TANHF(x) __builtin_amdgcn_tanhf(x)       // gfx1250 v_tanh_f32
#elif __has_builtin(__builtin_amdgcn_tanh_f32)
#define TANHF(x) __builtin_amdgcn_tanh_f32(x)
#else
#define TANHF(x) tanhf(x)
#endif

__global__ __launch_bounds__(BLOCK)
void ham_flow_wmma(const float* __restrict__ x0,
                   const float* __restrict__ W1,
                   const float* __restrict__ b1,
                   const float* __restrict__ W2,
                   float* __restrict__ out)
{
  // Both orientations of W1 resident in LDS f16 so ALL WMMA fragment loads are
  // contiguous 16-half runs (-> ds_load_b128). Total LDS ~= 280 KB < 320 KB/WGP;
  // with only 16 blocks in flight each block owns a WGP, so the LDS spend is free.
  __shared__ _Float16 sW1 [FEAT * PITCH];  // W1[f][h]   (backward B:  W1^T[k_h][f] = sW1[f][k_h])
  __shared__ _Float16 sW1T[HID  * PITCH];  // W1^T[h][f] (forward  B:  W1[k_f][h]  = sW1T[h][k_f])
  __shared__ _Float16 sQ  [TILE_M * PITCH]; // f16 eval-point q (16 samples x 256 feat)
  __shared__ _Float16 sU  [TILE_M * PITCH]; // f16 u = (1 - tanh^2 z) * W2

  const int tid  = threadIdx.x;
  const int lane = tid & 31;
  const int wave = tid >> 5;
  const int nlo  = lane & 15;
  const int hi16 = (lane >> 4) & 1;        // 0 = lanes 0-15, 1 = lanes 16-31
  const int nb   = blockIdx.x * TILE_M;    // first sample handled by this block

  // stage W1 -> LDS in both orientations (f16, padded rows); float4 global reads
  {
    const float4* W14 = (const float4*)W1;
    for (int i = tid; i < (FEAT * HID) / 4; i += BLOCK) {
      float4 v = W14[i];
      int f = (i * 4) >> 8, h = (i * 4) & 255;
      sW1 [f * PITCH + h + 0] = (_Float16)v.x;
      sW1 [f * PITCH + h + 1] = (_Float16)v.y;
      sW1 [f * PITCH + h + 2] = (_Float16)v.z;
      sW1 [f * PITCH + h + 3] = (_Float16)v.w;
      sW1T[(h + 0) * PITCH + f] = (_Float16)v.x;
      sW1T[(h + 1) * PITCH + f] = (_Float16)v.y;
      sW1T[(h + 2) * PITCH + f] = (_Float16)v.z;
      sW1T[(h + 3) * PITCH + f] = (_Float16)v.w;
    }
  }

  // per-lane constants for this wave's hidden tile (forward output column)
  const int   hcol = wave * 16 + nlo;
  const float b1v  = b1[hcol];
  const float w2v  = W2[hcol];

  // This thread owns the 8 state elements its WMMA D-fragment produces:
  //   m = r + 8*hi16 (sample row in tile), f = wave*16 + nlo (feature column)
  const int fown = wave * 16 + nlo;
  float xq[8], xp[8], yp[8], aq[8], ap[8];
  #pragma unroll
  for (int r = 0; r < 8; ++r) {
    int m = r + 8 * hi16;
    long idx = ((long)(nb + m) * FEAT + fown) * 2;  // x0[n, d, nn, {q,p}]
    xq[r] = x0[idx + 0];
    xp[r] = x0[idx + 1];
    yp[r] = xp[r];
    aq[r] = 0.f;
    ap[r] = 0.f;
    sQ[m * PITCH + fown] = (_Float16)xq[r];         // first eval point y1 = x
  }
  __syncthreads();

  // fragment K-offsets, contiguous order k = 16*hi16 + j (16-bit A 16x32, 16-bit B 32x16; A and B match)
  const int kA0  = hi16 ? 16 : 0;
  const int kB0  = hi16 ? 16 : 0;
  const int rowm = nlo;

  for (int step = 0; step < STEPS; ++step) {
    #pragma unroll 1
    for (int sub = 0; sub < 4; ++sub) {
      // ---------- forward: z(:, h-tile=wave) = qf @ W1 ----------
      v8f acc = {};
      #pragma unroll
      for (int kt = 0; kt < FEAT / 32; ++kt) {
        v16h A, B;
        const int ka = kt * 32 + kA0;
        #pragma unroll
        for (int j = 0; j < 16; ++j) {
          A[j]     = sQ[rowm * PITCH + ka + j];
        }
        const int kb = kt * 32 + kB0;
        #pragma unroll
        for (int j = 0; j < 16; ++j)
          B[j] = sW1T[hcol * PITCH + kb + j];        // W1[k_feat][hcol], contiguous
        acc = __builtin_amdgcn_wmma_f32_16x16x32_f16(false, A, false, B,
                                                     (short)0, acc, false, false);
      }
      // tanh' gate, scale by W2 -> u (f16 to LDS)
      #pragma unroll
      for (int r = 0; r < 8; ++r) {
        float z = acc[r] + b1v;
        float t = TANHF(z);
        float u = (1.f - t * t) * w2v;
        sU[(r + 8 * hi16) * PITCH + hcol] = (_Float16)u;
      }
      __syncthreads();

      // ---------- backward: g(:, f-tile=wave) = u @ W1^T ----------
      v8f g = {};
      #pragma unroll
      for (int kt = 0; kt < HID / 32; ++kt) {
        v16h A, B;
        const int ka = kt * 32 + kA0;
        #pragma unroll
        for (int j = 0; j < 16; ++j) {
          A[j]     = sU[rowm * PITCH + ka + j];
        }
        const int kb = kt * 32 + kB0;
        #pragma unroll
        for (int j = 0; j < 16; ++j)
          B[j] = sW1[fown * PITCH + kb + j];          // W1^T[k_hid][fown], contiguous
        g = __builtin_amdgcn_wmma_f32_16x16x32_f16(false, A, false, B,
                                                   (short)0, g, false, false);
      }

      // ---------- RK4 state update on owned elements (all in registers) ----------
      const float w = (sub == 0 || sub == 3) ? 1.f : 2.f;
      const float c = (sub == 2) ? DT : 0.5f * DT;
      #pragma unroll
      for (int r = 0; r < 8; ++r) {
        int m = r + 8 * hi16;
        float gv = g[r];                // dp/dt = -dH/dq
        aq[r] += w * yp[r];             // dq/dt = p at eval point
        ap[r] -= w * gv;
        float nq, np_;
        if (sub < 3) {                  // next eval point y = x + c*k
          nq  = xq[r] + c * yp[r];
          np_ = xp[r] - c * gv;
        } else {                        // x += (DT/6)*(k1+2k2+2k3+k4); y1' = x'
          nq  = xq[r] + (DT / 6.f) * aq[r];
          np_ = xp[r] + (DT / 6.f) * ap[r];
          xq[r] = nq;
          xp[r] = np_;
          aq[r] = 0.f;
          ap[r] = 0.f;
        }
        yp[r] = np_;
        sQ[m * PITCH + fown] = (_Float16)nq;
      }
      __syncthreads();
    }
  }

  // write final state back in [N, D, NN, 2] layout
  #pragma unroll
  for (int r = 0; r < 8; ++r) {
    int m = r + 8 * hi16;
    long idx = ((long)(nb + m) * FEAT + fown) * 2;
    out[idx + 0] = xq[r];
    out[idx + 1] = xp[r];
  }
}

extern "C" void kernel_launch(void* const* d_in, const int* in_sizes, int n_in,
                              void* d_out, int out_size, void* d_ws, size_t ws_size,
                              hipStream_t stream) {
  (void)in_sizes; (void)n_in; (void)d_ws; (void)ws_size; (void)out_size;
  const float* x0 = (const float*)d_in[0];
  const float* W1 = (const float*)d_in[1];
  const float* b1 = (const float*)d_in[2];
  const float* W2 = (const float*)d_in[3];
  // d_in[4] (b2) does not affect the vector field (gradient of a constant).
  ham_flow_wmma<<<NBATCH / TILE_M, BLOCK, 0, stream>>>(x0, W1, b1, W2, (float*)d_out);
}
